// GAT_51402168598675
// MI455X (gfx1250) — matrix operand order repaired and checked by reading, not compiled, not run on hardware
//
#include <hip/hip_runtime.h>
#include <hip/hip_bf16.h>

// ---------------------------------------------------------------------------
// GAT (3-layer) for MI455X / gfx1250.
// GEMMs: v_wmma_f32_16x16x32_f16. K and Nc templated -> fully unrolled k-loop
// with all addresses folded to base+immediate. Each wave owns a 16x64 (16x32)
// output strip: one A-fragment load feeds 4 (2) WMMAs. B tiles staged
// transposed through LDS: per-thread column strip gathered with immediate-
// offset 16-bit loads, written with one ds_store_b128, fragments read back as
// 2x ds_load_b128 per lane. Tail waves clamped -> branch-free WMMA body.
// Segment softmax + scatter aggregation use hardware global float atomics.
// ---------------------------------------------------------------------------

typedef __attribute__((ext_vector_type(16))) _Float16 v16h;
typedef __attribute__((ext_vector_type(8)))  _Float16 v8h;
typedef __attribute__((ext_vector_type(8)))  float    v8f;

#define GAT_N      50000
#define GAT_E0     800000
#define GAT_ET     850000      // E0 + N self loops
#define NEG_SLOPE  0.2f

// ------------------------- small helpers -----------------------------------

__device__ __forceinline__ void get_edge(const long long* __restrict__ ei, int e,
                                         int& s, int& d) {
  if (e < GAT_E0) { s = (int)ei[e]; d = (int)ei[GAT_E0 + e]; }
  else            { s = e - GAT_E0; d = e - GAT_E0; }
}

__device__ __forceinline__ void atomicMaxF(float* addr, float val) {
  int iv = __float_as_int(val);
  if (iv >= 0) atomicMax((int*)addr, iv);
  else         atomicMin((unsigned int*)addr, (unsigned int)iv);
}

__device__ __forceinline__ void atomAddF(float* p, float v) {
  unsafeAtomicAdd(p, v);   // lowers to global_atomic_add_f32 on gfx1250
}

// ------------------------- utility kernels ---------------------------------

__global__ void k_cvt_f16(const float* __restrict__ in, _Float16* __restrict__ out, int n) {
  int i = blockIdx.x * blockDim.x + threadIdx.x;
  if (i < n) out[i] = (_Float16)in[i];
}

__global__ void k_fill(float* __restrict__ p, float v, int n) {
  int i = blockIdx.x * blockDim.x + threadIdx.x;
  if (i < n) p[i] = v;
}

// ------------------------- WMMA GEMM ---------------------------------------
// C[M,Nc] = A16[M,K] @ B16[K,Nc], fp16 in / fp32 out.
// K in {128,256}, Nc in {256,32}, NT in {4,2} 16-wide column tiles per wave.
// grid: (ceil((M/16)/8), Nc/(NT*16)); block: 256 threads = 8 waves.
template <int K, int Nc, int NT>
__global__ void __launch_bounds__(256)
k_gemm_wmma(const _Float16* __restrict__ A, const _Float16* __restrict__ B,
            float* __restrict__ C, int M) {
  constexpr int COLS = NT * 16;       // columns staged per block
  constexpr int KPT  = 2 * NT;        // K-elements gathered per thread
  __shared__ __align__(64) _Float16 Bt[COLS * 32];   // [col][k] transposed

  const int lane   = threadIdx.x & 31;
  const int wave   = threadIdx.x >> 5;
  const int mTiles = M / 16;
  int tileM = blockIdx.x * 8 + wave;
  if (tileM >= mTiles) tileM = mTiles - 1;     // clamp tail: branch-free body
  const int colBase = blockIdx.y * COLS;

  // ---- per-thread B staging strip: column c, K-run kq..kq+KPT-1
  const int c  = threadIdx.x % COLS;
  const int kq = (threadIdx.x / COLS) * KPT;
  const _Float16* bstage = B + (size_t)kq * Nc + colBase + c;  // + k0*Nc per step
  _Float16* lstage = &Bt[c * 32 + kq];

  // ---- A-fragment addressing (contiguous k order, same as B fragment):
  // lanes 0-15: row = lane,    K = {0..15}
  // lanes16-31: row = lane-16, K = {16..31}
  const int arow  = tileM * 16 + (lane & 15);
  const int kbase = (lane >> 4) * 16;
  const _Float16* abase = A + (size_t)arow * K + kbase;        // + k0 per step

  // ---- B-fragment addressing: lane holds col = lane&15, contiguous 16 K
  const int bcol = lane & 15;
  const int bkh  = (lane >> 4) * 16;

  v8f acc[NT] = {};

#pragma unroll
  for (int k0 = 0; k0 < K; k0 += 32) {
    // stage B slab (32 x COLS) transposed into LDS: gather K-run of one
    // column (immediate offsets), write contiguous -> single wide ds store.
    {
      const _Float16* bp = bstage + (size_t)k0 * Nc;
      _Float16 vals[KPT];
#pragma unroll
      for (int i = 0; i < KPT; ++i) vals[i] = bp[(size_t)i * Nc];
      __builtin_memcpy(lstage, vals, sizeof(vals));  // ds_store_b128 / b64
    }
    __syncthreads();

    const _Float16* ap = abase + k0;
    if (k0 + 32 < K) __builtin_prefetch(ap + 32, 0, 1);   // global_prefetch_b8
    v8h alo = *(const v8h*)(ap);
    v8h ahi = *(const v8h*)(ap + 8);
    v16h afrag;
#pragma unroll
    for (int i = 0; i < 8; ++i) { afrag[i] = alo[i]; afrag[8 + i] = ahi[i]; }

#pragma unroll
    for (int j = 0; j < NT; ++j) {
      v16h bfrag = *(const v16h*)(&Bt[(j * 16 + bcol) * 32 + bkh]); // 2x ds_load_b128
      acc[j] = __builtin_amdgcn_wmma_f32_16x16x32_f16(
          /*neg_a=*/false, afrag, /*neg_b=*/false, bfrag,
          /*c_mod=*/(short)0, acc[j], /*reuse_a=*/false, /*reuse_b=*/false);
    }
    __syncthreads();
  }

  // D layout: VGPR r -> M = r (lanes 0-15) / r+8 (lanes 16-31), N = lane&15
  const int mro = (lane >> 4) * 8;
  float* cbase = C + (size_t)(tileM * 16 + mro) * Nc + colBase + bcol;
#pragma unroll
  for (int j = 0; j < NT; ++j) {
#pragma unroll
    for (int r = 0; r < 8; ++r) {
      cbase[(size_t)r * Nc + j * 16] = acc[j][r];
    }
  }
}

// ------------------------- attention kernels -------------------------------

// per (node, head): alpha_src / alpha_dst dot products
__global__ void k_node_alpha(const float* __restrict__ xl,
                             const float* __restrict__ av,
                             const float* __restrict__ dv,
                             float* __restrict__ asn, float* __restrict__ adn,
                             int H, int C) {
  int idx = blockIdx.x * blockDim.x + threadIdx.x;
  if (idx >= GAT_N * H) return;
  int n = idx / H, h = idx % H;
  const float* xr = xl + (size_t)n * H * C + (size_t)h * C;
  const float* a  = av + h * C;
  const float* d  = dv + h * C;
  float s1 = 0.f, s2 = 0.f;
  for (int c = 0; c < C; ++c) { float v = xr[c]; s1 += v * a[c]; s2 += v * d[c]; }
  asn[idx] = s1;
  adn[idx] = s2;
}

// pass 1: segment max over destination
__global__ void k_edge_max(const long long* __restrict__ ei,
                           const float* __restrict__ asn, const float* __restrict__ adn,
                           float* __restrict__ m, int H) {
  int idx = blockIdx.x * blockDim.x + threadIdx.x;
  if (idx >= GAT_ET * H) return;
  int e = idx / H, h = idx % H;
  int s, d; get_edge(ei, e, s, d);
  float v = asn[s * H + h] + adn[d * H + h];
  v = (v > 0.f) ? v : NEG_SLOPE * v;
  atomicMaxF(&m[d * H + h], v);
}

// pass 2: exp(e - max) and segment sum
__global__ void k_edge_expsum(const long long* __restrict__ ei,
                              const float* __restrict__ asn, const float* __restrict__ adn,
                              const float* __restrict__ m, float* __restrict__ ssum,
                              float* __restrict__ eexp, int H) {
  int idx = blockIdx.x * blockDim.x + threadIdx.x;
  if (idx >= GAT_ET * H) return;
  int e = idx / H, h = idx % H;
  int s, d; get_edge(ei, e, s, d);
  float v = asn[s * H + h] + adn[d * H + h];
  v = (v > 0.f) ? v : NEG_SLOPE * v;
  float ex = __expf(v - m[d * H + h]);
  eexp[idx] = ex;
  atomAddF(&ssum[d * H + h], ex);
}

// pass 3: alpha = e / sum; scatter-aggregate messages
__global__ void k_edge_agg(const long long* __restrict__ ei,
                           const float* __restrict__ eexp, const float* __restrict__ ssum,
                           const float* __restrict__ xl, float* __restrict__ agg,
                           float* __restrict__ alpha_out, int H, int C) {
  int idx = blockIdx.x * blockDim.x + threadIdx.x;
  if (idx >= GAT_ET * H) return;
  int e = idx / H, h = idx % H;
  int s, d; get_edge(ei, e, s, d);
  float alpha = eexp[idx] / (ssum[d * H + h] + 1e-16f);
  alpha_out[idx] = alpha;
  const float* xr = xl + (size_t)s * H * C + (size_t)h * C;
  float* ar = agg + (size_t)d * H * C + (size_t)h * C;
  for (int c = 0; c < C; ++c) atomAddF(&ar[c], xr[c] * alpha);
}

// bias + ELU (concat layers)
__global__ void k_bias_elu(const float* __restrict__ agg, const float* __restrict__ bias,
                           float* __restrict__ out, int F, int n) {
  int i = blockIdx.x * blockDim.x + threadIdx.x;
  if (i >= n) return;
  float v = agg[i] + bias[i % F];
  out[i] = (v > 0.f) ? v : (__expf(v) - 1.f);
}

// bias only (final layer, heads==1 so mean == identity)
__global__ void k_bias_out(const float* __restrict__ agg, const float* __restrict__ bias,
                           float* __restrict__ out, int F, int n) {
  int i = blockIdx.x * blockDim.x + threadIdx.x;
  if (i >= n) return;
  out[i] = agg[i] + bias[i % F];
}

// write the (src;dst) self-loop-augmented edge index (as float) three times
__global__ void k_write_ei(const long long* __restrict__ ei,
                           float* __restrict__ o1, float* __restrict__ o2,
                           float* __restrict__ o3) {
  int e = blockIdx.x * blockDim.x + threadIdx.x;
  if (e >= GAT_ET) return;
  int s, d; get_edge(ei, e, s, d);
  float fs = (float)s, fd = (float)d;
  o1[e] = fs; o1[GAT_ET + e] = fd;
  o2[e] = fs; o2[GAT_ET + e] = fd;
  o3[e] = fs; o3[GAT_ET + e] = fd;
}

// ------------------------- layer driver (host) ------------------------------

static void run_layer(const float* feat, int Fin, const float* W,
                      const float* av, const float* dv, const float* bias,
                      int H, int C, const long long* ei,
                      float* xl, float* agg, _Float16* a16, _Float16* w16,
                      float* asn, float* adn, float* mbuf, float* sbuf, float* eexp,
                      float* alpha_out, float* feat_out, bool concat,
                      hipStream_t stream) {
  const int F = H * C;
  const int TPB = 256;

  int nA = GAT_N * Fin;
  k_cvt_f16<<<(nA + TPB - 1) / TPB, TPB, 0, stream>>>(feat, a16, nA);
  int nB = Fin * F;
  k_cvt_f16<<<(nB + TPB - 1) / TPB, TPB, 0, stream>>>(W, w16, nB);

  const int mBlocks = (GAT_N / 16 + 7) / 8;
  if (Fin == 128 && F == 256) {
    dim3 gg(mBlocks, 256 / 64);
    k_gemm_wmma<128, 256, 4><<<gg, TPB, 0, stream>>>(a16, w16, xl, GAT_N);
  } else if (Fin == 256 && F == 256) {
    dim3 gg(mBlocks, 256 / 64);
    k_gemm_wmma<256, 256, 4><<<gg, TPB, 0, stream>>>(a16, w16, xl, GAT_N);
  } else { // Fin == 256, F == 32
    dim3 gg(mBlocks, 1);
    k_gemm_wmma<256, 32, 2><<<gg, TPB, 0, stream>>>(a16, w16, xl, GAT_N);
  }

  int nh = GAT_N * H;
  k_node_alpha<<<(nh + TPB - 1) / TPB, TPB, 0, stream>>>(xl, av, dv, asn, adn, H, C);
  k_fill<<<(nh + TPB - 1) / TPB, TPB, 0, stream>>>(mbuf, -__builtin_huge_valf(), nh);
  k_fill<<<(nh + TPB - 1) / TPB, TPB, 0, stream>>>(sbuf, 0.f, nh);
  int nagg = GAT_N * F;
  k_fill<<<(nagg + TPB - 1) / TPB, TPB, 0, stream>>>(agg, 0.f, nagg);

  int ne = GAT_ET * H;
  k_edge_max   <<<(ne + TPB - 1) / TPB, TPB, 0, stream>>>(ei, asn, adn, mbuf, H);
  k_edge_expsum<<<(ne + TPB - 1) / TPB, TPB, 0, stream>>>(ei, asn, adn, mbuf, sbuf, eexp, H);
  k_edge_agg   <<<(ne + TPB - 1) / TPB, TPB, 0, stream>>>(ei, eexp, sbuf, xl, agg, alpha_out, H, C);

  if (concat)
    k_bias_elu<<<(nagg + TPB - 1) / TPB, TPB, 0, stream>>>(agg, bias, feat_out, F, nagg);
  else
    k_bias_out<<<(nagg + TPB - 1) / TPB, TPB, 0, stream>>>(agg, bias, feat_out, F, nagg);
}

// ------------------------- entry point --------------------------------------

extern "C" void kernel_launch(void* const* d_in, const int* in_sizes, int n_in,
                              void* d_out, int out_size, void* d_ws, size_t ws_size,
                              hipStream_t stream) {
  const float*     x   = (const float*)d_in[0];
  const long long* ei  = (const long long*)d_in[1];
  const float*     W1  = (const float*)d_in[2];
  const float*     as1 = (const float*)d_in[3];
  const float*     ad1 = (const float*)d_in[4];
  const float*     b1  = (const float*)d_in[5];
  const float*     W2  = (const float*)d_in[6];
  const float*     as2 = (const float*)d_in[7];
  const float*     ad2 = (const float*)d_in[8];
  const float*     b2  = (const float*)d_in[9];
  const float*     W3  = (const float*)d_in[10];
  const float*     as3 = (const float*)d_in[11];
  const float*     ad3 = (const float*)d_in[12];
  const float*     b3  = (const float*)d_in[13];
  float* out = (float*)d_out;

  // flat output offsets (floats): out, (ei,a1), (ei,a2), (ei,a3)
  const size_t OUT0 = 0;
  const size_t EI1  = OUT0 + (size_t)GAT_N * 32;          //  1,600,000
  const size_t A1o  = EI1 + 2ull * GAT_ET;                //  3,300,000
  const size_t EI2  = A1o + (size_t)GAT_ET * 4;           //  6,700,000
  const size_t A2o  = EI2 + 2ull * GAT_ET;                //  8,400,000
  const size_t EI3  = A2o + (size_t)GAT_ET * 4;           // 11,800,000
  const size_t A3o  = EI3 + 2ull * GAT_ET;                // 13,500,000

  // workspace carve (256B aligned bump allocator)
  char* wp = (char*)d_ws;
  auto alloc = [&](size_t bytes) -> void* {
    void* p = (void*)wp;
    wp += (bytes + 255) & ~(size_t)255;
    return p;
  };
  float*    xl   = (float*)alloc((size_t)GAT_N * 256 * 4);
  float*    agg  = (float*)alloc((size_t)GAT_N * 256 * 4);
  float*    h1   = (float*)alloc((size_t)GAT_N * 256 * 4);
  float*    h2   = (float*)alloc((size_t)GAT_N * 256 * 4);
  _Float16* a16  = (_Float16*)alloc((size_t)GAT_N * 256 * 2);
  _Float16* w16  = (_Float16*)alloc((size_t)256 * 256 * 2);
  float*    asn  = (float*)alloc((size_t)GAT_N * 4 * 4);
  float*    adn  = (float*)alloc((size_t)GAT_N * 4 * 4);
  float*    mbuf = (float*)alloc((size_t)GAT_N * 4 * 4);
  float*    sbuf = (float*)alloc((size_t)GAT_N * 4 * 4);
  float*    eexp = (float*)alloc((size_t)GAT_ET * 4 * 4);

  k_write_ei<<<(GAT_ET + 255) / 256, 256, 0, stream>>>(ei, out + EI1, out + EI2, out + EI3);

  // layer 1: 128 -> 4x64 (concat 256), ELU
  run_layer(x, 128, W1, as1, ad1, b1, 4, 64, ei,
            xl, agg, a16, w16, asn, adn, mbuf, sbuf, eexp,
            out + A1o, h1, /*concat=*/true, stream);
  // layer 2: 256 -> 4x64 (concat 256), ELU
  run_layer(h1, 256, W2, as2, ad2, b2, 4, 64, ei,
            xl, agg, a16, w16, asn, adn, mbuf, sbuf, eexp,
            out + A2o, h2, /*concat=*/true, stream);
  // layer 3: 256 -> 1x32 (mean over 1 head == identity), no activation
  run_layer(h2, 256, W3, as3, ad3, b3, 1, 32, ei,
            xl, agg, a16, w16, asn, adn, mbuf, sbuf, eexp,
            out + A3o, out + OUT0, /*concat=*/false, stream);

  (void)in_sizes; (void)n_in; (void)out_size; (void)ws_size;
}
